// MultiHeadAttentionClassical_65481071399213
// MI455X (gfx1250) — matrix operand order repaired and checked by reading, not compiled, not run on hardware
//
#include <hip/hip_runtime.h>
#include <stdint.h>

// MultiHeadAttention for MI455X (gfx1250), bf16 WMMA with fp32 accumulation.
// Pipeline: fp32->bf16 convert, QKV GEMMs (v_wmma_f32_16x16x32_bf16, async-LDS
// double-buffered), flash-attention (online softmax, async K prefetch,
// register-prefetched V transpose), output GEMM.
// Workspace layout (assumes ws_size >= 48 MB):
//   [0,8M)   xb bf16 [4096][1024]
//   [8,16M)  Wq/Wk/Wv/Wo bf16 (2MB each)
//   [16,40M) Q,K,V bf16 [B][H][S][D] (8MB each)
//   [40,48M) ctx bf16 [B][S][E]

#define BATCHN 2
#define SEQN   2048
#define EMBN   1024
#define HEADS  16
#define HDIM   64

typedef __attribute__((ext_vector_type(16))) __bf16 v16bf;
typedef __attribute__((ext_vector_type(8)))  float  v8f;

union Frag16 { uint4 u[2]; v16bf v; };

__device__ inline v8f wmma_bf16(v16bf a, v16bf b, v8f c) {
  // D = A(16x32) * B(32x16) + C, fp32 accumulate
  return __builtin_amdgcn_wmma_f32_16x16x32_bf16(false, a, false, b,
                                                 (short)0, c, false, false);
}

__device__ inline v8f zero8() {
  v8f z;
  #pragma unroll
  for (int i = 0; i < 8; ++i) z[i] = 0.f;
  return z;
}

// ---- async global -> LDS copy (CDNA5 GLOBAL_LOAD_ASYNC_TO_LDS_B128) ----
#if __has_builtin(__builtin_amdgcn_global_load_async_to_lds_b128)
#define HAVE_ASYNC_LDS 1
#else
#define HAVE_ASYNC_LDS 0
#endif

typedef int v4i_vs __attribute__((__vector_size__(16)));
typedef __attribute__((address_space(1))) v4i_vs* gbl_v4i;
typedef __attribute__((address_space(3))) v4i_vs* lds_v4i;

__device__ __forceinline__ void cp_async16(void* lds, const void* g) {
#if HAVE_ASYNC_LDS
  // LDS offset is the low 32 bits of the generic pointer; global VA is the
  // flat address itself. Build AS-qualified pointers via integers (no
  // addrspacecast needed).
  __builtin_amdgcn_global_load_async_to_lds_b128(
      (gbl_v4i)(uintptr_t)g,
      (lds_v4i)(uint32_t)(uintptr_t)lds,
      0, 0);
#else
  *(uint4*)lds = *(const uint4*)g;
#endif
}

__device__ __forceinline__ void async_wait0() {
#if HAVE_ASYNC_LDS
#if __has_builtin(__builtin_amdgcn_s_wait_asynccnt)
  __builtin_amdgcn_s_wait_asynccnt(0);
#else
  asm volatile("s_wait_asynccnt 0" ::: "memory");
#endif
#endif
}

// ---------------- fp32 -> bf16 convert ----------------
__global__ void cvt_kernel(const float* __restrict__ s, __bf16* __restrict__ d, int n) {
  int i = (blockIdx.x * 256 + threadIdx.x) * 4;
  if (i >= n) return;
  float4 f = *(const float4*)(s + i);
  __align__(8) __bf16 t[4] = {(__bf16)f.x, (__bf16)f.y, (__bf16)f.z, (__bf16)f.w};
  *(uint2*)(d + i) = *(const uint2*)t;
}

// ---------------- GEMM: C[m,n] = sum_k A[m,k]*W[n,k] + bias[n] ----------------
// A: [M][K] bf16 row-major, W: [N][K] bf16 row-major (== B^T, B-frags read W rows).
// mode 0: store bf16 into [B][H][S][D]; mode 1: store fp32 row-major [M][N].
// Double-buffered 128x128x32 tiles staged with async LDS copies.
#define GTM 128
#define GTN 128
#define GTK 32
#define GLS (GTK + 8)  // 40 elements = 80B row stride, keeps 16B alignment

__global__ void __launch_bounds__(256)
gemm_bf16_nt(const __bf16* __restrict__ A, const __bf16* __restrict__ W,
             const float* __restrict__ bias, void* __restrict__ outp,
             int Mdim, int Ndim, int Kdim, int mode) {
  __shared__ __bf16 sA[2][GTM][GLS];
  __shared__ __bf16 sB[2][GTN][GLS];
  const int tid  = threadIdx.x;
  const int lane = tid & 31, wave = tid >> 5;
  const int l16  = lane & 15, lh = lane >> 4, h8 = lh * 8;
  const int wm   = wave >> 2, wn = wave & 3;   // 2(M) x 4(N) waves, 64x32 per wave
  const int m0   = blockIdx.y * GTM;
  const int n0   = blockIdx.x * GTN;

  auto stage = [&](int st, int kb) {
    #pragma unroll
    for (int i = 0; i < 2; ++i) {
      int c = tid + 256 * i;          // 512 chunks (8 bf16) per matrix
      int r = c >> 2, cc = c & 3;     // 4 chunks per 32-wide row
      cp_async16(&sA[st][r][cc * 8], A + (size_t)(m0 + r) * Kdim + kb + cc * 8);
      cp_async16(&sB[st][r][cc * 8], W + (size_t)(n0 + r) * Kdim + kb + cc * 8);
    }
  };

  v8f acc[4][2];
  #pragma unroll
  for (int i = 0; i < 4; ++i)
    #pragma unroll
    for (int j = 0; j < 2; ++j) acc[i][j] = zero8();

  stage(0, 0);
  async_wait0();
  __syncthreads();

  const int nkb = Kdim / GTK;
  for (int ib = 0; ib < nkb; ++ib) {
    const int cur = ib & 1;
    if (ib + 1 < nkb) stage(cur ^ 1, (ib + 1) * GTK);

    Frag16 af[4];
    #pragma unroll
    for (int bd = 0; bd < 4; ++bd) {
      const __bf16* ar = &sA[cur][wm * 64 + bd * 16 + l16][h8];
      af[bd].u[0] = *(const uint4*)ar;
      af[bd].u[1] = *(const uint4*)(ar + 16);
    }
    Frag16 bfr[2];
    #pragma unroll
    for (int t = 0; t < 2; ++t) {
      const __bf16* br = &sB[cur][wn * 32 + t * 16 + l16][h8];
      bfr[t].u[0] = *(const uint4*)br;
      bfr[t].u[1] = *(const uint4*)(br + 16);
    }
    #pragma unroll
    for (int bd = 0; bd < 4; ++bd)
      #pragma unroll
      for (int t = 0; t < 2; ++t)
        acc[bd][t] = wmma_bf16(af[bd].v, bfr[t].v, acc[bd][t]);

    async_wait0();
    __syncthreads();
  }

  #pragma unroll
  for (int t = 0; t < 2; ++t) {
    int n = n0 + wn * 32 + t * 16 + l16;
    float bv = bias[n];
    #pragma unroll
    for (int bd = 0; bd < 4; ++bd) {
      #pragma unroll
      for (int j = 0; j < 8; ++j) {
        int m = m0 + wm * 64 + bd * 16 + h8 + j;
        float v = acc[bd][t][j] + bv;
        if (mode == 0) {
          int bb = m >> 11, s = m & (SEQN - 1);
          int hh = n >> 6, d = n & (HDIM - 1);
          ((__bf16*)outp)[(((size_t)(bb * HEADS + hh)) * SEQN + s) * HDIM + d] = (__bf16)v;
        } else {
          ((float*)outp)[(size_t)m * Ndim + n] = v;
        }
      }
    }
  }
}

// ---------------- Flash attention ----------------
// Grid: (S/64, B*H). Block: 128 threads (4 waves), each wave owns 16 q-rows.
// K tiles prefetched via async LDS copies, V tiles prefetched into registers
// and transposed into the next LDS stage after the compute phase.
__global__ void __launch_bounds__(128)
flash_attn(const __bf16* __restrict__ Q, const __bf16* __restrict__ K,
           const __bf16* __restrict__ V, const int* __restrict__ mask,
           __bf16* __restrict__ ctx) {
  __shared__ __bf16 sK[2][64][64];    // [key][d] row-major
  __shared__ __bf16 sVt[2][64][64];   // [d][key] (transposed V)
  __shared__ __bf16 sP[4][16][64];    // per-wave P layout fix-up scratch

  const int qt = blockIdx.x, bh = blockIdx.y;
  const int b = bh >> 4, h = bh & 15;
  const int tid  = threadIdx.x;
  const int lane = tid & 31, wave = tid >> 5;
  const int l16  = lane & 15, lh = lane >> 4, h8 = lh * 8;

  const __bf16* kbase = K + (size_t)bh * SEQN * HDIM;
  const __bf16* vbase = V + (size_t)bh * SEQN * HDIM;

  uint4 vr0[2], vr1[2];
  const int kp = tid & 31, dgB = tid >> 5;

  auto loadKasync = [&](int st, int kt) {
    const uint4* kg = (const uint4*)(kbase + (size_t)kt * 64 * HDIM);
    uint4* dst = (uint4*)&sK[st][0][0];
    #pragma unroll
    for (int i = 0; i < 4; ++i) {
      int c = tid + 128 * i;
      cp_async16(dst + c, kg + c);
    }
  };
  auto loadVregs = [&](int kt) {
    const __bf16* vg = vbase + (size_t)kt * 64 * HDIM;
    #pragma unroll
    for (int g = 0; g < 2; ++g) {
      int d0 = (dgB * 2 + g) * 8;
      vr0[g] = *(const uint4*)(vg + (kp * 2) * HDIM + d0);
      vr1[g] = *(const uint4*)(vg + (kp * 2 + 1) * HDIM + d0);
    }
  };
  auto storeVt = [&](int st) {
    #pragma unroll
    for (int g = 0; g < 2; ++g) {
      int d0 = (dgB * 2 + g) * 8;
      const __bf16* e0 = (const __bf16*)&vr0[g];
      const __bf16* e1 = (const __bf16*)&vr1[g];
      #pragma unroll
      for (int e = 0; e < 8; ++e) {
        __align__(4) __bf16 pair[2] = { e0[e], e1[e] };
        *(uint32_t*)&sVt[st][d0 + e][kp * 2] = *(const uint32_t*)pair;
      }
    }
  };

  // Q fragments (A-layout), rows = qt*64 + wave*16 + l16, kept in regs
  const __bf16* qrow = Q + ((size_t)bh * SEQN + qt * 64 + wave * 16 + l16) * HDIM;
  Frag16 q0, q1;
  q0.u[0] = *(const uint4*)(qrow + h8);
  q0.u[1] = *(const uint4*)(qrow + 16 + h8);
  q1.u[0] = *(const uint4*)(qrow + 32 + h8);
  q1.u[1] = *(const uint4*)(qrow + 48 + h8);

  float mrow[8], lrow[8];
  v8f acc[4];
  #pragma unroll
  for (int j = 0; j < 8; ++j) { mrow[j] = -3.0e38f; lrow[j] = 0.f; }
  #pragma unroll
  for (int t = 0; t < 4; ++t) acc[t] = zero8();

  // prologue: stage tile 0
  loadKasync(0, 0);
  loadVregs(0);
  storeVt(0);
  async_wait0();
  __syncthreads();

  const int NKT = SEQN / 64;
  for (int kt = 0; kt < NKT; ++kt) {
    const int cur = kt & 1;

    // mask for this tile first, so its loadcnt wait doesn't drain V prefetch
    int mk[4];
    #pragma unroll
    for (int t = 0; t < 4; ++t) mk[t] = mask[b * SEQN + kt * 64 + t * 16 + l16];

    if (kt + 1 < NKT) {
      loadKasync(cur ^ 1, kt + 1);
      loadVregs(kt + 1);
    }

    // S = Q * K^T  (B-frag of K^T == rows of K)
    v8f s[4];
    #pragma unroll
    for (int t = 0; t < 4; ++t) {
      const __bf16* kr = &sK[cur][t * 16 + l16][0];
      Frag16 b0, b1;
      b0.u[0] = *(const uint4*)(kr + h8);
      b0.u[1] = *(const uint4*)(kr + 16 + h8);
      b1.u[0] = *(const uint4*)(kr + 32 + h8);
      b1.u[1] = *(const uint4*)(kr + 48 + h8);
      v8f c = zero8();
      c = wmma_bf16(q0.v, b0.v, c);
      c = wmma_bf16(q1.v, b1.v, c);
      s[t] = c;
    }

    // scale + mask
    #pragma unroll
    for (int t = 0; t < 4; ++t)
      #pragma unroll
      for (int j = 0; j < 8; ++j) {
        float v = s[t][j] * 0.125f;       // 1/sqrt(64)
        if (mk[t] == 0) v = -1.0e9f;
        s[t][j] = v;
      }

    // row max (rows span 16 lanes in C/D layout -> butterfly over masks 8..1)
    float rm[8];
    #pragma unroll
    for (int j = 0; j < 8; ++j)
      rm[j] = fmaxf(fmaxf(s[0][j], s[1][j]), fmaxf(s[2][j], s[3][j]));
    #pragma unroll
    for (int off = 8; off >= 1; off >>= 1)
      #pragma unroll
      for (int j = 0; j < 8; ++j)
        rm[j] = fmaxf(rm[j], __shfl_xor(rm[j], off, 32));

    // online softmax update
    float al[8], rs[8];
    #pragma unroll
    for (int j = 0; j < 8; ++j) {
      float mn = fmaxf(mrow[j], rm[j]);
      al[j] = __expf(mrow[j] - mn);
      mrow[j] = mn;
      float sum = 0.f;
      #pragma unroll
      for (int t = 0; t < 4; ++t) {
        float p = __expf(s[t][j] - mn);
        s[t][j] = p;
        sum += p;
      }
      rs[j] = sum;
    }
    #pragma unroll
    for (int off = 8; off >= 1; off >>= 1)
      #pragma unroll
      for (int j = 0; j < 8; ++j)
        rs[j] += __shfl_xor(rs[j], off, 32);
    #pragma unroll
    for (int j = 0; j < 8; ++j) lrow[j] = lrow[j] * al[j] + rs[j];
    #pragma unroll
    for (int t = 0; t < 4; ++t)
      #pragma unroll
      for (int j = 0; j < 8; ++j) acc[t][j] *= al[j];

    // P: D-layout -> LDS row-major -> reload as A-fragments (per-wave scratch)
    #pragma unroll
    for (int t = 0; t < 4; ++t)
      #pragma unroll
      for (int j = 0; j < 8; ++j)
        sP[wave][h8 + j][t * 16 + l16] = (__bf16)s[t][j];

    asm volatile("s_wait_dscnt 0" ::: "memory");

    const __bf16* pw = &sP[wave][l16][0];
    Frag16 pa0, pa1;
    pa0.u[0] = *(const uint4*)(pw + h8);
    pa0.u[1] = *(const uint4*)(pw + 16 + h8);
    pa1.u[0] = *(const uint4*)(pw + 32 + h8);
    pa1.u[1] = *(const uint4*)(pw + 48 + h8);

    // acc += P * V  (B-frag of V == rows of V^T == sVt rows)
    #pragma unroll
    for (int t = 0; t < 4; ++t) {
      const __bf16* vr = &sVt[cur][t * 16 + l16][0];
      Frag16 b0, b1;
      b0.u[0] = *(const uint4*)(vr + h8);
      b0.u[1] = *(const uint4*)(vr + 16 + h8);
      b1.u[0] = *(const uint4*)(vr + 32 + h8);
      b1.u[1] = *(const uint4*)(vr + 48 + h8);
      acc[t] = wmma_bf16(pa0.v, b0.v, acc[t]);
      acc[t] = wmma_bf16(pa1.v, b1.v, acc[t]);
    }

    if (kt + 1 < NKT) storeVt(cur ^ 1);

    async_wait0();
    __syncthreads();
  }

  // normalize and store ctx[b][s][h*64+d] as bf16
  float inv[8];
  #pragma unroll
  for (int j = 0; j < 8; ++j) inv[j] = 1.f / lrow[j];
  #pragma unroll
  for (int t = 0; t < 4; ++t)
    #pragma unroll
    for (int j = 0; j < 8; ++j) {
      int srow = qt * 64 + wave * 16 + h8 + j;
      int col  = h * HDIM + t * 16 + l16;
      ctx[((size_t)b * SEQN + srow) * EMBN + col] = (__bf16)(acc[t][j] * inv[j]);
    }
}

// ---------------- host side ----------------
extern "C" void kernel_launch(void* const* d_in, const int* in_sizes, int n_in,
                              void* d_out, int out_size, void* d_ws, size_t ws_size,
                              hipStream_t stream) {
  (void)in_sizes; (void)n_in; (void)out_size; (void)ws_size;
  const float* x  = (const float*)d_in[0];
  const int*   mk = (const int*)  d_in[1];
  const float* Wq = (const float*)d_in[2];
  const float* bq = (const float*)d_in[3];
  const float* Wk = (const float*)d_in[4];
  const float* bk = (const float*)d_in[5];
  const float* Wv = (const float*)d_in[6];
  const float* bv = (const float*)d_in[7];
  const float* Wo = (const float*)d_in[8];
  const float* bo = (const float*)d_in[9];

  char* ws = (char*)d_ws;
  const size_t MB = (size_t)1 << 20;
  __bf16* xb  = (__bf16*)(ws + 0 * MB);
  __bf16* wqb = (__bf16*)(ws + 8 * MB);
  __bf16* wkb = (__bf16*)(ws + 10 * MB);
  __bf16* wvb = (__bf16*)(ws + 12 * MB);
  __bf16* wob = (__bf16*)(ws + 14 * MB);
  __bf16* Qb  = (__bf16*)(ws + 16 * MB);
  __bf16* Kb  = (__bf16*)(ws + 24 * MB);
  __bf16* Vb  = (__bf16*)(ws + 32 * MB);
  __bf16* ctx = (__bf16*)(ws + 40 * MB);

  const int nx = BATCHN * SEQN * EMBN;  // 4194304
  const int nw = EMBN * EMBN;           // 1048576
  cvt_kernel<<<nx / 1024, 256, 0, stream>>>(x,  xb,  nx);
  cvt_kernel<<<nw / 1024, 256, 0, stream>>>(Wq, wqb, nw);
  cvt_kernel<<<nw / 1024, 256, 0, stream>>>(Wk, wkb, nw);
  cvt_kernel<<<nw / 1024, 256, 0, stream>>>(Wv, wvb, nw);
  cvt_kernel<<<nw / 1024, 256, 0, stream>>>(Wo, wob, nw);

  dim3 gg(EMBN / GTN, (BATCHN * SEQN) / GTM);  // (8, 32)
  gemm_bf16_nt<<<gg, 256, 0, stream>>>(xb, wqb, bq, Qb, BATCHN * SEQN, EMBN, EMBN, 0);
  gemm_bf16_nt<<<gg, 256, 0, stream>>>(xb, wkb, bk, Kb, BATCHN * SEQN, EMBN, EMBN, 0);
  gemm_bf16_nt<<<gg, 256, 0, stream>>>(xb, wvb, bv, Vb, BATCHN * SEQN, EMBN, EMBN, 0);

  flash_attn<<<dim3(SEQN / 64, BATCHN * HEADS), 128, 0, stream>>>(Qb, Kb, Vb, mk, ctx);

  gemm_bf16_nt<<<gg, 256, 0, stream>>>(ctx, wob, bo, d_out, BATCHN * SEQN, EMBN, EMBN, 1);
}
